// GraphAttentionLayer_81784767250584
// MI455X (gfx1250) — matrix operand order repaired and checked by reading, not compiled, not run on hardware
//
#include <hip/hip_runtime.h>
#include <hip/hip_bf16.h>
#include <math.h>

typedef __attribute__((ext_vector_type(16))) _Float16 v16h;
typedef __attribute__((ext_vector_type(8)))  _Float16 v8h;
typedef __attribute__((ext_vector_type(8)))  float    v8f;

#define GB 8     // batch
#define GN 1024  // tokens
#define GD 128   // d_model
#define GH 8     // heads
#define GHD 16   // head dim

// ---------------------------------------------------------------------------
// Kernel 0: pack the four 128x128 f32 weight matrices into f16 laid out in the
// exact WMMA B-matrix VGPR order:  Wp[mz][kk][ct][lane][j] with
//   K row  = kk*32 + (j>>3)*16 + (lane>>4)*8 + (j&7),   column = ct*16 + (lane&15).
// Each B tile is then one aligned 32-byte v16h load per lane, no converts.
// ---------------------------------------------------------------------------
__global__ __launch_bounds__(256) void gat_pack_w(
    const float* __restrict__ Wq, const float* __restrict__ Wk,
    const float* __restrict__ Wv, const float* __restrict__ Wo,
    _Float16* __restrict__ Wp)
{
    const int mz = blockIdx.x;  // 0=Q 1=K 2=V 3=O
    const float* W = (mz == 0) ? Wq : (mz == 1) ? Wk : (mz == 2) ? Wv : Wo;
    _Float16* dst = Wp + (size_t)mz * GD * GD;
    for (int o = threadIdx.x; o < GD * GD; o += 256) {
        const int j    = o & 15;
        const int lane = (o >> 4) & 31;
        const int ct   = (o >> 9) & 7;
        const int kk   = o >> 12;
        const int krow = kk * 32 + (j >> 3) * 16 + (lane >> 4) * 8 + (j & 7);
        const int c    = ct * 16 + (lane & 15);
        dst[o] = (_Float16)W[(size_t)krow * GD + c];
    }
}

__device__ __forceinline__ v16h load_a_f32(const float* __restrict__ xrow, int k0) {
    // A tile halves: j<8 -> K=k0+j ; j>=8 -> K=k0+16+(j&7)  (16B-aligned float4s)
    const float4* p0 = (const float4*)(xrow + k0);
    const float4* p1 = (const float4*)(xrow + k0 + 16);
    const float4 f0 = p0[0], f1 = p0[1], f2 = p1[0], f3 = p1[1];
    v16h a;
    a[0]  = (_Float16)f0.x; a[1]  = (_Float16)f0.y;
    a[2]  = (_Float16)f0.z; a[3]  = (_Float16)f0.w;
    a[4]  = (_Float16)f1.x; a[5]  = (_Float16)f1.y;
    a[6]  = (_Float16)f1.z; a[7]  = (_Float16)f1.w;
    a[8]  = (_Float16)f2.x; a[9]  = (_Float16)f2.y;
    a[10] = (_Float16)f2.z; a[11] = (_Float16)f2.w;
    a[12] = (_Float16)f3.x; a[13] = (_Float16)f3.y;
    a[14] = (_Float16)f3.z; a[15] = (_Float16)f3.w;
    return a;
}

// ---------------------------------------------------------------------------
// Kernel 1: fused QKV projection.  x (f32) @ Wp-f16 + b.
// grid = (N/16, B, 3); block = 32 (1 wave).  16x128 strip per wave.
// Q,K stored f16 [b][h][n][16]; V stored transposed [b][h][16][N].
// ---------------------------------------------------------------------------
__global__ __launch_bounds__(32) void gat_qkv(
    const float* __restrict__ x, const _Float16* __restrict__ Wp,
    const float* __restrict__ bq, const float* __restrict__ bk,
    const float* __restrict__ bv,
    _Float16* __restrict__ Qh, _Float16* __restrict__ Kh,
    _Float16* __restrict__ Vt)
{
    const int lane = threadIdx.x;
    const int nt   = blockIdx.x;
    const int b    = blockIdx.y;
    const int mz   = blockIdx.z;           // 0=Q 1=K 2=V
    const _Float16* Wsel = Wp + (size_t)mz * GD * GD;
    const float*    bias = (mz == 0) ? bq : (mz == 1) ? bk : bv;

    const int m  = lane & 15;
    const int hi = lane >> 4;
    const int n0 = nt * 16;

    v8f acc[8] = {};
    const float* xrow = x + ((size_t)b * GN + n0 + m) * GD;

    for (int kk = 0; kk < 4; ++kk) {
        const v16h a = load_a_f32(xrow, kk * 32 + hi * 8);
        #pragma unroll
        for (int ct = 0; ct < 8; ++ct) {
            const v16h bm = *(const v16h*)(
                Wsel + ((size_t)(kk * 8 + ct) * 32 + lane) * 16);
            acc[ct] = __builtin_amdgcn_wmma_f32_16x16x32_f16(
                false, a, false, bm, (short)0, acc[ct], false, false);
        }
    }

    #pragma unroll
    for (int ct = 0; ct < 8; ++ct) {
        const int c  = ct * 16 + m;        // out column = h*16 + d
        const float bb = bias[c];
        #pragma unroll
        for (int v = 0; v < 8; ++v) {
            const float val = acc[ct][v] + bb;
            const int n = n0 + v + 8 * hi; // C layout: row = v + 8*hi
            if (mz == 0)
                Qh[(((size_t)b * GH + ct) * GN + n) * GHD + m] = (_Float16)val;
            else if (mz == 1)
                Kh[(((size_t)b * GH + ct) * GN + n) * GHD + m] = (_Float16)val;
            else
                Vt[(((size_t)b * GH + ct) * GHD + m) * GN + n] = (_Float16)val;
        }
    }
}

// ---------------------------------------------------------------------------
// Kernel 2: fused edge-biased masked flash attention.
// grid = (N/16, B); block = 256 (8 waves, wave w == head w).
// edge_weights / adj_mask tile staged in LDS once per key tile for all heads.
// ---------------------------------------------------------------------------
__global__ __launch_bounds__(256) void gat_attn(
    const _Float16* __restrict__ Qh, const _Float16* __restrict__ Kh,
    const _Float16* __restrict__ Vt,
    const float* __restrict__ ew, const int* __restrict__ adj,
    const float* __restrict__ We, const float* __restrict__ be,
    _Float16* __restrict__ Oh)
{
    __shared__ float    ew_s[16 * 32];
    __shared__ int      adj_s[16 * 32];
    __shared__ _Float16 P_s[GH][16 * 32];

    const int tid  = threadIdx.x;
    const int lane = tid & 31;
    const int h    = tid >> 5;             // wave id == head
    const int qt   = blockIdx.x;
    const int b    = blockIdx.y;
    const int q0   = qt * 16;
    const int m    = lane & 15;
    const int hi   = lane >> 4;

    const float wE = We[h];
    const float bE = be[h];

    // Q tile as A matrix, K padded 16->32 with zeros (hd = 16).
    v16h aq = {};
    if (hi == 0) {
        aq = *(const v16h*)(
            Qh + (((size_t)b * GH + h) * GN + q0 + m) * GHD);
    }

    v8f   oacc = {};
    float mrun[8], lrun[8];
    #pragma unroll
    for (int v = 0; v < 8; ++v) { mrun[v] = -1e30f; lrun[v] = 0.0f; }

    const int NKT = GN / 32;               // 32 key tiles of 32 keys
    for (int kt = 0; kt < NKT; ++kt) {
        const int kbase = kt * 32;
        __syncthreads();                   // prior-iter LDS readers done
        {
            const int i2 = tid * 2;        // 512 elems, exact coverage
            const int r = i2 >> 5, cc = i2 & 31;
            const size_t gi = ((size_t)b * GN + q0 + r) * GN + kbase + cc;
            *(float2*)&ew_s[i2]  = *(const float2*)&ew[gi];
            *(int2*)&adj_s[i2]   = *(const int2*)&adj[gi];
        }
        if (tid < 16 && kt + 1 < NKT) {    // hint next edge tile -> cache
            const size_t gn = ((size_t)b * GN + q0 + tid) * GN + kbase + 32;
            __builtin_prefetch(&ew[gn], 0, 0);
            __builtin_prefetch(&adj[gn], 0, 0);
        }
        __syncthreads();

        // B tiles for scores: K^T, keys kbase..+15 and +16..+31.
        // lane col = key = m; K index = d = hi*16+j -> hi==1 half is zero pad.
        v16h bk0 = {}, bk1 = {};
        if (hi == 0) {
            const _Float16* kr0 =
                Kh + (((size_t)b * GH + h) * GN + kbase + m) * GHD;
            bk0 = *(const v16h*)kr0;
            bk1 = *(const v16h*)(kr0 + 16 * GHD);
        }
        v8f z = {};
        v8f s0 = __builtin_amdgcn_wmma_f32_16x16x32_f16(false, aq, false, bk0,
                                                        (short)0, z, false, false);
        v8f s1 = __builtin_amdgcn_wmma_f32_16x16x32_f16(false, aq, false, bk1,
                                                        (short)0, z, false, false);

        float x0[8], x1[8], tv[8];
        #pragma unroll
        for (int v = 0; v < 8; ++v) {
            const int r = v + 8 * hi;      // query row of this element
            float a0 = s0[v] * 0.25f + ew_s[r * 32 + m]      * wE + bE;
            float a1 = s1[v] * 0.25f + ew_s[r * 32 + 16 + m] * wE + bE;
            if (adj_s[r * 32 + m]      == 0) a0 = -__builtin_inff();
            if (adj_s[r * 32 + 16 + m] == 0) a1 = -__builtin_inff();
            x0[v] = a0; x1[v] = a1;
            tv[v] = fmaxf(a0, a1);
        }
        // row max: reduce across the 16 lanes holding the same row
        #pragma unroll
        for (int v = 0; v < 8; ++v) {
            float t = tv[v];
            t = fmaxf(t, __shfl_xor(t, 1));
            t = fmaxf(t, __shfl_xor(t, 2));
            t = fmaxf(t, __shfl_xor(t, 4));
            t = fmaxf(t, __shfl_xor(t, 8));
            tv[v] = t;
        }
        #pragma unroll
        for (int v = 0; v < 8; ++v) {
            const int r = v + 8 * hi;
            const float mn = fmaxf(mrun[v], tv[v]);
            const float sc = __expf(mrun[v] - mn);
            const float p0 = __expf(x0[v] - mn);
            const float p1 = __expf(x1[v] - mn);
            lrun[v] = lrun[v] * sc + p0 + p1;   // per-lane partial row sum
            mrun[v] = mn;
            oacc[v] = oacc[v] * sc;
            P_s[h][r * 32 + m]      = (_Float16)p0;
            P_s[h][r * 32 + 16 + m] = (_Float16)p1;
        }
        __builtin_amdgcn_wave_barrier();   // same-wave LDS RAW: DS is in-order

        // P back as A matrix (16x32, full K) via two ds_load_b128
        const v8h plo = *(const v8h*)&P_s[h][m * 32 + hi * 16];
        const v8h phi = *(const v8h*)&P_s[h][m * 32 + hi * 16 + 8];
        const v16h ap = __builtin_shufflevector(plo, phi, 0, 1, 2, 3, 4, 5, 6, 7,
                                                8, 9, 10, 11, 12, 13, 14, 15);
        // V as B matrix: contiguous 32B per lane thanks to the Vt transpose
        const v16h bv_ = *(const v16h*)(
            Vt + (((size_t)b * GH + h) * GHD + m) * GN + kbase + hi * 16);
        oacc = __builtin_amdgcn_wmma_f32_16x16x32_f16(false, ap, false, bv_,
                                                      (short)0, oacc, false, false);
    }

    // finalize: divide by full row sum, store f16 head-major output
    #pragma unroll
    for (int v = 0; v < 8; ++v) {
        float l = lrun[v];
        l += __shfl_xor(l, 1);
        l += __shfl_xor(l, 2);
        l += __shfl_xor(l, 4);
        l += __shfl_xor(l, 8);
        const float o = oacc[v] / l;
        const int n = q0 + v + 8 * hi;
        Oh[(((size_t)b * GH + h) * GN + n) * GHD + m] = (_Float16)o;
    }
}

// ---------------------------------------------------------------------------
// Kernel 3: output projection.  Oh (f16, head-major) @ Wo-f16 + bo -> f32.
// grid = (N/16, B); block = 32 (1 wave); 16x128 strip per wave.
// ---------------------------------------------------------------------------
__global__ __launch_bounds__(32) void gat_out(
    const _Float16* __restrict__ Oh, const _Float16* __restrict__ WpO,
    const float* __restrict__ bo, float* __restrict__ out)
{
    const int lane = threadIdx.x;
    const int nt = blockIdx.x;
    const int b  = blockIdx.y;
    const int m  = lane & 15;
    const int hi = lane >> 4;
    const int n0 = nt * 16;

    v8f acc[8] = {};
    for (int kk = 0; kk < 4; ++kk) {
        // A: concat dim kd = h*16+d; halves j<8 -> h=2kk, j>=8 -> h=2kk+1,
        // d = hi*8 + (j&7): two aligned v8h runs in Oh.
        const v8h a0 = *(const v8h*)(
            Oh + (((size_t)b * GH + kk * 2)     * GN + n0 + m) * GHD + hi * 8);
        const v8h a1 = *(const v8h*)(
            Oh + (((size_t)b * GH + kk * 2 + 1) * GN + n0 + m) * GHD + hi * 8);
        const v16h a = __builtin_shufflevector(a0, a1, 0, 1, 2, 3, 4, 5, 6, 7,
                                               8, 9, 10, 11, 12, 13, 14, 15);
        #pragma unroll
        for (int ct = 0; ct < 8; ++ct) {
            const v16h bm = *(const v16h*)(
                WpO + ((size_t)(kk * 8 + ct) * 32 + lane) * 16);
            acc[ct] = __builtin_amdgcn_wmma_f32_16x16x32_f16(
                false, a, false, bm, (short)0, acc[ct], false, false);
        }
    }
    #pragma unroll
    for (int ct = 0; ct < 8; ++ct) {
        const int c = ct * 16 + m;
        const float bb = bo[c];
        #pragma unroll
        for (int v = 0; v < 8; ++v)
            out[((size_t)b * GN + n0 + v + 8 * hi) * GD + c] = acc[ct][v] + bb;
    }
}

// ---------------------------------------------------------------------------
extern "C" void kernel_launch(void* const* d_in, const int* in_sizes, int n_in,
                              void* d_out, int out_size, void* d_ws, size_t ws_size,
                              hipStream_t stream) {
    const float* x   = (const float*)d_in[0];
    const int*   adj = (const int*)d_in[1];
    const float* ew  = (const float*)d_in[2];
    const float* Wq  = (const float*)d_in[3];
    const float* bq  = (const float*)d_in[4];
    const float* Wk  = (const float*)d_in[5];
    const float* bk  = (const float*)d_in[6];
    const float* Wv  = (const float*)d_in[7];
    const float* bv  = (const float*)d_in[8];
    const float* Wo  = (const float*)d_in[9];
    const float* bo  = (const float*)d_in[10];
    const float* We  = (const float*)d_in[11];
    const float* be  = (const float*)d_in[12];
    float* out = (float*)d_out;

    const size_t sz = (size_t)GB * GH * GN * GHD;   // 1M f16 elems per tensor
    _Float16* Qh = (_Float16*)d_ws;
    _Float16* Kh = Qh + sz;
    _Float16* Vt = Kh + sz;
    _Float16* Oh = Vt + sz;
    _Float16* Wp = Oh + sz;                         // 4 * 16384 f16 = 128 KB

    gat_pack_w<<<dim3(4), 256, 0, stream>>>(Wq, Wk, Wv, Wo, Wp);

    dim3 g1(GN / 16, GB, 3);
    gat_qkv<<<g1, 32, 0, stream>>>(x, Wp, bq, bk, bv, Qh, Kh, Vt);

    dim3 g2(GN / 16, GB);
    gat_attn<<<g2, 256, 0, stream>>>(Qh, Kh, Vt, ew, adj, We, be, Oh);

    dim3 g3(GN / 16, GB);
    gat_out<<<g3, 32, 0, stream>>>(Oh, Wp + (size_t)3 * GD * GD, bo, out);
}
